// TransformerGRUAttnCRF_31920196944620
// MI455X (gfx1250) — matrix operand order repaired and checked by reading, not compiled, not run on hardware
//
#include <hip/hip_runtime.h>
#include <hip/hip_bf16.h>
#include <math.h>

// Problem constants (match reference)
#define Vv   32000
#define Ee   512
#define Hh   256
#define Dd   512
#define DFFc 2048
#define NBc  4
#define Cc   16
#define Bc   32
#define Tc   512
#define Mr   (Bc * Tc)   // 16384 rows

typedef __bf16 bf16;
typedef __attribute__((ext_vector_type(16))) __bf16 v16bf;
typedef __attribute__((ext_vector_type(8)))  float  v8f;

union FragB { v16bf v; bf16 e[16]; };

// A-fragment (16x32 bf16, M x K): lane<16 row=lane K=k0..k0+15,
// lane>=16 row=lane-16 K=k0+16..k0+31  (matches load_fragB k order)
__device__ __forceinline__ v16bf load_fragA(const bf16* rowbase, int ld, int lane, int k0) {
  FragB f;
  const int row = lane & 15;
  const int ko  = k0 + ((lane & 16) ? 16 : 0);
  const bf16* p = rowbase + (size_t)row * ld + ko;
#pragma unroll
  for (int i = 0; i < 16; ++i) f.e[i] = p[i];
  return f.v;
}

// B-fragment (32x16 bf16, K x N), stored as Wt[N][K] row-major:
// lane<16 col=lane K=k0..k0+15 ; lane>=16 col=lane-16 K=k0+16..k0+31
__device__ __forceinline__ v16bf load_fragB(const bf16* rowbase, int ld, int lane, int k0) {
  FragB f;
  const int col = lane & 15;
  const int ko  = k0 + ((lane & 16) ? 16 : 0);
  const bf16* p = rowbase + (size_t)col * ld + ko;
#pragma unroll
  for (int i = 0; i < 16; ++i) f.e[i] = p[i];
  return f.v;
}

// CDNA5 async global->LDS copy (16B per lane), tracked by ASYNCcnt.
// The LDS *pointer* is passed through a ptrtoint so the shared array is
// captured -> the asm "memory" clobber aliases it and the subsequent
// ds_load fragment gathers cannot be folded away. Low 32 bits of the
// generic pointer are the LDS byte address (ISA 10.2 aperture mapping).
__device__ __forceinline__ void async_ld_b128(const bf16* lds_ptr, const void* gaddr) {
  const unsigned lds_addr = (unsigned)(unsigned long long)lds_ptr;
  asm volatile("global_load_async_to_lds_b128 %0, %1, off"
               :: "v"(lds_addr), "v"(gaddr) : "memory");
}
__device__ __forceinline__ void wait_async_le3() { asm volatile("s_wait_asynccnt 0x3" ::: "memory"); }
__device__ __forceinline__ void wait_async_0()   { asm volatile("s_wait_asynccnt 0x0" ::: "memory"); }

// ---------------------------------------------------------------------------
// Tiled GEMM:  C[M,N] = A[M,K](bf16) x Wt[N,K]^T(bf16) + bias, optional ReLU,
// optional f32 and/or bf16 stores.
// Block = 256 thr (8 waves). Block tile 128x64, wave tile 32x32 (2x2 WMMA).
// Double-buffered LDS fed by global_load_async_to_lds_b128 (ASYNCcnt).
// M multiple of 128, K multiple of 32; N guarded (N=16 emission GEMM).
// ---------------------------------------------------------------------------
#define GSTG 7680                // halves per stage: A 128*40 + B 64*40
__global__ __launch_bounds__(256)
void gemm_bf16(const bf16* __restrict__ A, const bf16* __restrict__ Wt,
               const float* __restrict__ bias, float* __restrict__ Cf,
               bf16* __restrict__ Cb, int Mm, int Nn, int Kk, int relu)
{
  __shared__ bf16 smem[2 * GSTG];   // 30 KB, double buffered

  const int tid  = threadIdx.x;
  const int lane = tid & 31;
  const int wv   = tid >> 5;          // 0..7
  const int mw   = wv & 3;            // 32-row slab within 128
  const int nw   = wv >> 2;           // 0..1 -> 32-col slab within 64

  const int m0 = blockIdx.y * 128;
  const int n0 = blockIdx.x * 64;

  // per-thread copy assignment: 3 x 16B chunks per stage
  const int rr  = tid >> 2;           // 0..63
  const int sub = (tid & 3) * 8;      // half offset within a 32-half row
  int nB = n0 + rr; if (nB >= Nn) nB = Nn - 1;   // clamp (keep ASYNCcnt uniform)
  const bf16* gA0 = A  + (size_t)(m0 + rr)      * Kk + sub;
  const bf16* gA1 = A  + (size_t)(m0 + 64 + rr) * Kk + sub;
  const bf16* gB  = Wt + (size_t)nB             * Kk + sub;
  bf16* lA0 = smem + (rr * 40 + sub);
  bf16* lA1 = smem + ((64 + rr) * 40 + sub);
  bf16* lB  = smem + (5120 + rr * 40 + sub);

  v8f acc00 = {}, acc01 = {}, acc10 = {}, acc11 = {};

  const int nk = Kk >> 5;
  // prologue: stage 0
  async_ld_b128(lA0, gA0);
  async_ld_b128(lA1, gA1);
  async_ld_b128(lB,  gB);

  for (int kk = 0; kk < nk; ++kk) {
    const int cur = kk & 1;
    if (kk + 1 < nk) {               // issue stage kk+1 into the other buffer
      const int koff = (kk + 1) << 5;
      const int st   = ((kk + 1) & 1) * GSTG;     // halves
      async_ld_b128(lA0 + st, gA0 + koff);
      async_ld_b128(lA1 + st, gA1 + koff);
      async_ld_b128(lB  + st, gB  + koff);
      wait_async_le3();              // stage kk complete (older 3 retired)
    } else {
      wait_async_0();
    }
    __syncthreads();                 // all waves' stage-kk data visible

    const bf16* sa = &smem[cur * GSTG];
    const bf16* sb = &smem[cur * GSTG + 5120];
    v16bf a0 = load_fragA(sa + (mw * 32) * 40,      40, lane, 0);
    v16bf a1 = load_fragA(sa + (mw * 32 + 16) * 40, 40, lane, 0);
    v16bf b0 = load_fragB(sb + (nw * 32) * 40,      40, lane, 0);
    v16bf b1 = load_fragB(sb + (nw * 32 + 16) * 40, 40, lane, 0);
    acc00 = __builtin_amdgcn_wmma_f32_16x16x32_bf16(false, a0, false, b0, (short)0, acc00, false, false);
    acc01 = __builtin_amdgcn_wmma_f32_16x16x32_bf16(false, a0, false, b1, (short)0, acc01, false, false);
    acc10 = __builtin_amdgcn_wmma_f32_16x16x32_bf16(false, a1, false, b0, (short)0, acc10, false, false);
    acc11 = __builtin_amdgcn_wmma_f32_16x16x32_bf16(false, a1, false, b1, (short)0, acc11, false, false);
    __syncthreads();                 // stage-kk buffer free for reuse
  }

  // epilogue: C layout lanes0-15 -> M=r, lanes16-31 -> M=8+r ; N = base+lane%16
  const int mb = m0 + mw * 32;
  const int nb = n0 + nw * 32;
#pragma unroll
  for (int i = 0; i < 2; ++i) {
    const v8f& aL = i ? acc10 : acc00;
    const v8f& aR = i ? acc11 : acc01;
#pragma unroll
    for (int r = 0; r < 8; ++r) {
      const int m  = mb + i * 16 + ((lane & 16) ? (r + 8) : r);
      const int nA = nb + (lane & 15);
      const int nBo = nA + 16;
      if (nA < Nn) {
        float v = aL[r] + (bias ? bias[nA] : 0.0f);
        if (relu) v = fmaxf(v, 0.0f);
        if (Cf) Cf[(size_t)m * Nn + nA] = v;
        if (Cb) Cb[(size_t)m * Nn + nA] = (bf16)v;
      }
      if (nBo < Nn) {
        float v = aR[r] + (bias ? bias[nBo] : 0.0f);
        if (relu) v = fmaxf(v, 0.0f);
        if (Cf) Cf[(size_t)m * Nn + nBo] = v;
        if (Cb) Cb[(size_t)m * Nn + nBo] = (bf16)v;
      }
    }
  }
}

// ---------------------------------------------------------------------------
// Bi-GRU sequential scan. One persistent workgroup per direction (grid=2),
// 512 threads = 16 waves. Hidden state (32x256) resident in LDS (f32 + bf16).
// Per step: gh = h @ w_hh^T via 96 WMMA tiles (16x16); each wave owns matched
// (r,z,n) gate tiles for its j-columns so gate math stays in-register.
// ---------------------------------------------------------------------------
__global__ __launch_bounds__(512)
void gru_scan(const bf16* __restrict__ whhF, const bf16* __restrict__ whhB,
              const float* __restrict__ bhhF, const float* __restrict__ bhhB,
              const float* __restrict__ giF,  const float* __restrict__ giB,
              float* __restrict__ hsF,        float* __restrict__ hsB)
{
  const int dir = blockIdx.x;
  const bf16*  whh = dir ? whhB : whhF;
  const float* bhh = dir ? bhhB : bhhF;
  const float* gi  = dir ? giB  : giF;
  float*       hs  = dir ? hsB  : hsF;

  __shared__ float hF[32 * 256];   // 32 KB
  __shared__ bf16  hB[32 * 272];   // 17 KB (padded stride)

  const int tid  = threadIdx.x;
  const int lane = tid & 31;
  const int wv   = tid >> 5;        // 0..15
  const int mt   = wv & 1;          // batch-row slab (16 rows)
  const int jt0  = wv >> 1;         // j tile 0..7 (second tile = jt0+8)

  for (int i = tid; i < 32 * 256; i += 512) hF[i] = 0.0f;
  for (int i = tid; i < 32 * 272; i += 512) hB[i] = (bf16)0.0f;
  __syncthreads();

  for (int step = 0; step < Tc; ++step) {
    const int t = dir ? (Tc - 1 - step) : step;

    v8f aR0 = {}, aZ0 = {}, aN0 = {};
    v8f aR1 = {}, aZ1 = {}, aN1 = {};

#pragma unroll 1
    for (int kk = 0; kk < 8; ++kk) {
      const int k0 = kk * 32;
      v16bf af = load_fragA(&hB[(mt * 16) * 272], 272, lane, k0);
      {
        const int j0 = jt0 * 16;
        v16bf bR = load_fragB(whh + (size_t)(0   + j0) * 256, 256, lane, k0);
        v16bf bZ = load_fragB(whh + (size_t)(256 + j0) * 256, 256, lane, k0);
        v16bf bN = load_fragB(whh + (size_t)(512 + j0) * 256, 256, lane, k0);
        aR0 = __builtin_amdgcn_wmma_f32_16x16x32_bf16(false, af, false, bR, (short)0, aR0, false, false);
        aZ0 = __builtin_amdgcn_wmma_f32_16x16x32_bf16(false, af, false, bZ, (short)0, aZ0, false, false);
        aN0 = __builtin_amdgcn_wmma_f32_16x16x32_bf16(false, af, false, bN, (short)0, aN0, false, false);
      }
      {
        const int j1 = (jt0 + 8) * 16;
        v16bf bR = load_fragB(whh + (size_t)(0   + j1) * 256, 256, lane, k0);
        v16bf bZ = load_fragB(whh + (size_t)(256 + j1) * 256, 256, lane, k0);
        v16bf bN = load_fragB(whh + (size_t)(512 + j1) * 256, 256, lane, k0);
        aR1 = __builtin_amdgcn_wmma_f32_16x16x32_bf16(false, af, false, bR, (short)0, aR1, false, false);
        aZ1 = __builtin_amdgcn_wmma_f32_16x16x32_bf16(false, af, false, bZ, (short)0, aZ1, false, false);
        aN1 = __builtin_amdgcn_wmma_f32_16x16x32_bf16(false, af, false, bN, (short)0, aN1, false, false);
      }
    }
    __syncthreads();   // all reads of hB complete before updates

#pragma unroll 2
    for (int half = 0; half < 2; ++half) {
      const int jt = half ? (jt0 + 8) : jt0;
      const v8f& aR = half ? aR1 : aR0;
      const v8f& aZ = half ? aZ1 : aZ0;
      const v8f& aN = half ? aN1 : aN0;
#pragma unroll
      for (int r = 0; r < 8; ++r) {
        const int m = mt * 16 + ((lane & 16) ? (r + 8) : r);   // batch index
        const int j = jt * 16 + (lane & 15);
        const float* gib = gi + ((size_t)m * Tc + t) * 768;
        const float ir = gib[j], iz = gib[256 + j], in_ = gib[512 + j];
        const float hr = aR[r] + bhh[j];
        const float hz = aZ[r] + bhh[256 + j];
        const float hn = aN[r] + bhh[512 + j];
        const float rg = 1.0f / (1.0f + __expf(-(ir + hr)));
        const float zg = 1.0f / (1.0f + __expf(-(iz + hz)));
        const float ng = tanhf(in_ + rg * hn);
        const float ho = hF[m * 256 + j];
        const float hnew = (1.0f - zg) * ng + zg * ho;
        hF[m * 256 + j] = hnew;
        hB[m * 272 + j] = (bf16)hnew;
        hs[((size_t)m * Tc + t) * 256 + j] = hnew;
      }
    }
    __syncthreads();   // updates visible before next step's fragment loads
  }
}

// ---------------------------------------------------------------------------
// Elementwise / small kernels
// ---------------------------------------------------------------------------
__global__ void f2b(const float* __restrict__ in, bf16* __restrict__ out, size_t n) {
  size_t i = (size_t)blockIdx.x * blockDim.x + threadIdx.x;
  if (i < n) out[i] = (bf16)in[i];
}

__global__ void f2b_T(const float* __restrict__ in, bf16* __restrict__ out, int R, int Cn) {
  size_t i = (size_t)blockIdx.x * blockDim.x + threadIdx.x;
  if (i < (size_t)R * Cn) {
    int r = (int)(i / Cn), c = (int)(i % Cn);
    out[(size_t)c * R + r] = (bf16)in[i];
  }
}

__global__ void embed_pe(const int* __restrict__ x, const float* __restrict__ embW,
                         bf16* __restrict__ out) {
  size_t idx = (size_t)blockIdx.x * blockDim.x + threadIdx.x;
  if (idx >= (size_t)Mr * Ee) return;
  const int d = (int)(idx & (Ee - 1));
  const size_t bt = idx >> 9;                 // Ee == 512
  const int t = (int)(bt & (Tc - 1));         // Tc == 512
  const int tok = x[bt];
  const float div = __expf(-__logf(10000.0f) * (float)(d & ~1) / (float)Ee);
  const float ang = (float)t * div;
  const float pe  = (d & 1) ? __cosf(ang) : __sinf(ang);
  out[idx] = (bf16)(embW[(size_t)tok * Ee + d] + pe);
}

__global__ void concat_hf_hb(const float* __restrict__ hf, const float* __restrict__ hb,
                             float* __restrict__ hc, bf16* __restrict__ hcb) {
  size_t idx = (size_t)blockIdx.x * blockDim.x + threadIdx.x;
  if (idx >= (size_t)Mr * Dd) return;
  const int d = (int)(idx & (Dd - 1));
  const size_t bt = idx >> 9;
  const float v = (d < Hh) ? hf[bt * Hh + d] : hb[bt * Hh + (d - Hh)];
  hc[idx] = v;
  hcb[idx] = (bf16)v;
}

__global__ __launch_bounds__(256)
void resid_ln(float* __restrict__ h, bf16* __restrict__ hb, const float* __restrict__ ff,
              const float* __restrict__ g, const float* __restrict__ b) {
  __shared__ float red[256];
  const size_t row = blockIdx.x;
  const int tid = threadIdx.x;
  float v0 = h[row * Dd + tid]       + ff[row * Dd + tid];
  float v1 = h[row * Dd + 256 + tid] + ff[row * Dd + 256 + tid];

  red[tid] = v0 + v1; __syncthreads();
  for (int s = 128; s > 0; s >>= 1) { if (tid < s) red[tid] += red[tid + s]; __syncthreads(); }
  const float mu = red[0] / (float)Dd;
  __syncthreads();
  red[tid] = (v0 - mu) * (v0 - mu) + (v1 - mu) * (v1 - mu); __syncthreads();
  for (int s = 128; s > 0; s >>= 1) { if (tid < s) red[tid] += red[tid + s]; __syncthreads(); }
  const float rstd = rsqrtf(red[0] / (float)Dd + 1e-5f);

  const float o0 = (v0 - mu) * rstd * g[tid]       + b[tid];
  const float o1 = (v1 - mu) * rstd * g[256 + tid] + b[256 + tid];
  h[row * Dd + tid]        = o0;  hb[row * Dd + tid]        = (bf16)o0;
  h[row * Dd + 256 + tid]  = o1;  hb[row * Dd + 256 + tid]  = (bf16)o1;
}

// ---------------------------------------------------------------------------
// CRF Viterbi: single block, 512 thr = (b in 0..31) x (cur tag 0..15)
// ---------------------------------------------------------------------------
__global__ __launch_bounds__(512)
void crf_viterbi(const float* __restrict__ logits, const int* __restrict__ y,
                 const float* __restrict__ trans, int* __restrict__ bps,
                 float* __restrict__ out) {
  __shared__ float tr[Cc * Cc];
  __shared__ float sc[Bc * Cc];
  __shared__ float fin[Bc * Cc];
  __shared__ int   lastTag[Bc];

  const int tid = threadIdx.x;
  const int b = tid >> 4, c = tid & 15;
  if (tid < Cc * Cc) tr[tid] = trans[tid];
  __syncthreads();

  sc[tid] = tr[1 * Cc + c] + logits[((size_t)b * Tc + 0) * Cc + c];   // BOS=1
  __syncthreads();

  for (int t = 1; t < Tc; ++t) {
    const float e = logits[((size_t)b * Tc + t) * Cc + c];
    const int mask = (y[b * Tc + t] != 0);
    float best = -3.4e38f; int bp = 0;
#pragma unroll
    for (int p = 0; p < Cc; ++p) {
      const float v = sc[b * Cc + p] + tr[p * Cc + c];
      if (v > best) { best = v; bp = p; }
    }
    best += e;
    const float nw = mask ? best : sc[tid];
    bps[(size_t)(t - 1) * (Bc * Cc) + tid] = bp;
    __syncthreads();
    sc[tid] = nw;
    __syncthreads();
  }

  fin[tid] = sc[tid] + tr[c * Cc + 2];   // EOS=2
  __syncthreads();
  if (c == 0) {
    float bv = -3.4e38f; int bt_ = 0;
    for (int cc = 0; cc < Cc; ++cc) {
      const float v = fin[b * Cc + cc];
      if (v > bv) { bv = v; bt_ = cc; }
    }
    out[b] = bv; lastTag[b] = bt_;
  }
  __syncthreads();
  if (tid < Bc) {
    int tag = lastTag[tid];
    float* path = out + Bc + (size_t)tid * Tc;
    for (int s = Tc - 2; s >= 0; --s) {
      path[s + 1] = (float)tag;
      const int prev = bps[(size_t)s * (Bc * Cc) + tid * Cc + tag];
      if (y[tid * Tc + s + 1] != 0) tag = prev;
    }
    path[0] = (float)tag;
  }
}

// ---------------------------------------------------------------------------
extern "C" void kernel_launch(void* const* d_in, const int* in_sizes, int n_in,
                              void* d_out, int out_size, void* d_ws, size_t ws_size,
                              hipStream_t stream) {
  const int*   x      = (const int*)  d_in[0];
  const int*   y      = (const int*)  d_in[1];
  const float* embW   = (const float*)d_in[2];
  const float* w_ih_f = (const float*)d_in[3];
  const float* w_hh_f = (const float*)d_in[4];
  const float* b_ih_f = (const float*)d_in[5];
  const float* b_hh_f = (const float*)d_in[6];
  const float* w_ih_b = (const float*)d_in[7];
  const float* w_hh_b = (const float*)d_in[8];
  const float* b_ih_b = (const float*)d_in[9];
  const float* b_hh_b = (const float*)d_in[10];
  const float* ff_w1  = (const float*)d_in[11];
  const float* ff_b1  = (const float*)d_in[12];
  const float* ff_w2  = (const float*)d_in[13];
  const float* ff_b2  = (const float*)d_in[14];
  const float* ln_g   = (const float*)d_in[15];
  const float* ln_b   = (const float*)d_in[16];
  const float* fc_w   = (const float*)d_in[17];
  const float* fc_b   = (const float*)d_in[18];
  const float* trans  = (const float*)d_in[19];

  char* p = (char*)d_ws;
  auto take = [&](size_t bytes) -> char* {
    char* r = p; p += (bytes + 255) & ~(size_t)255; return r;
  };
  bf16*  xb    = (bf16*) take((size_t)Mr * Ee * 2);
  bf16*  wihfB = (bf16*) take((size_t)3 * Hh * Ee * 2);
  bf16*  wihbB = (bf16*) take((size_t)3 * Hh * Ee * 2);
  bf16*  whhfB = (bf16*) take((size_t)3 * Hh * Hh * 2);
  bf16*  whhbB = (bf16*) take((size_t)3 * Hh * Hh * 2);
  bf16*  w1t   = (bf16*) take((size_t)NBc * DFFc * Dd * 2);
  bf16*  w2t   = (bf16*) take((size_t)NBc * Dd * DFFc * 2);
  bf16*  fcwt  = (bf16*) take((size_t)Cc * Dd * 2);
  float* giF   = (float*)take((size_t)Mr * 3 * Hh * 4);
  float* giB   = (float*)take((size_t)Mr * 3 * Hh * 4);
  float* hsF   = (float*)take((size_t)Mr * Hh * 4);
  float* hsB   = (float*)take((size_t)Mr * Hh * 4);
  float* hcat  = (float*)take((size_t)Mr * Dd * 4);
  bf16*  hcatb = (bf16*) take((size_t)Mr * Dd * 2);
  bf16*  hidb  = (bf16*) take((size_t)Mr * DFFc * 2);
  float* ffout = (float*)take((size_t)Mr * Dd * 4);
  float* logit = (float*)take((size_t)Mr * Cc * 4);
  int*   bps   = (int*)  take((size_t)(Tc - 1) * Bc * Cc * 4);
  (void)ws_size; (void)n_in; (void)in_sizes; (void)out_size;

  // 1. weight conversion (+ transposition where needed)
  f2b<<<(3 * Hh * Ee + 255) / 256, 256, 0, stream>>>(w_ih_f, wihfB, (size_t)3 * Hh * Ee);
  f2b<<<(3 * Hh * Ee + 255) / 256, 256, 0, stream>>>(w_ih_b, wihbB, (size_t)3 * Hh * Ee);
  f2b<<<(3 * Hh * Hh + 255) / 256, 256, 0, stream>>>(w_hh_f, whhfB, (size_t)3 * Hh * Hh);
  f2b<<<(3 * Hh * Hh + 255) / 256, 256, 0, stream>>>(w_hh_b, whhbB, (size_t)3 * Hh * Hh);
  for (int i = 0; i < NBc; ++i) {
    f2b_T<<<(Dd * DFFc + 255) / 256, 256, 0, stream>>>(
        ff_w1 + (size_t)i * Dd * DFFc, w1t + (size_t)i * DFFc * Dd, Dd, DFFc);
    f2b_T<<<(DFFc * Dd + 255) / 256, 256, 0, stream>>>(
        ff_w2 + (size_t)i * DFFc * Dd, w2t + (size_t)i * Dd * DFFc, DFFc, Dd);
  }
  f2b_T<<<(Dd * Cc + 255) / 256, 256, 0, stream>>>(fc_w, fcwt, Dd, Cc);

  // 2. embedding + positional encoding -> bf16
  embed_pe<<<(Mr * Ee + 255) / 256, 256, 0, stream>>>(x, embW, xb);

  // 3. GRU input projections (one big GEMM per direction), bias folded in
  gemm_bf16<<<dim3(3 * Hh / 64, Mr / 128), 256, 0, stream>>>(
      xb, wihfB, b_ih_f, giF, nullptr, Mr, 3 * Hh, Ee, 0);
  gemm_bf16<<<dim3(3 * Hh / 64, Mr / 128), 256, 0, stream>>>(
      xb, wihbB, b_ih_b, giB, nullptr, Mr, 3 * Hh, Ee, 0);

  // 4. sequential bidirectional GRU (persistent workgroup per direction)
  gru_scan<<<2, 512, 0, stream>>>(whhfB, whhbB, b_hh_f, b_hh_b, giF, giB, hsF, hsB);

  // 5. concat -> (B,T,512) f32 + bf16
  concat_hf_hb<<<(Mr * Dd + 255) / 256, 256, 0, stream>>>(hsF, hsB, hcat, hcatb);

  // 6. FeedForward blocks (residual + LayerNorm)
  for (int i = 0; i < NBc; ++i) {
    gemm_bf16<<<dim3(DFFc / 64, Mr / 128), 256, 0, stream>>>(
        hcatb, w1t + (size_t)i * DFFc * Dd, ff_b1 + (size_t)i * DFFc,
        nullptr, hidb, Mr, DFFc, Dd, 1);                       // bias + ReLU -> bf16
    gemm_bf16<<<dim3(Dd / 64, Mr / 128), 256, 0, stream>>>(
        hidb, w2t + (size_t)i * Dd * DFFc, ff_b2 + (size_t)i * Dd,
        ffout, nullptr, Mr, Dd, DFFc, 0);
    resid_ln<<<Mr, 256, 0, stream>>>(hcat, hcatb, ffout,
                                     ln_g + (size_t)i * Dd, ln_b + (size_t)i * Dd);
  }

  // 7. emission logits (N=16, guarded tiles)
  gemm_bf16<<<dim3(1, Mr / 128), 256, 0, stream>>>(
      hcatb, fcwt, fc_b, logit, nullptr, Mr, Cc, Dd, 0);

  // 8. CRF Viterbi decode -> d_out = [best_score(32) | path(32x512)]
  crf_viterbi<<<1, 512, 0, stream>>>(logit, y, trans, bps, (float*)d_out);
}
